// SAttention_29583734735456
// MI455X (gfx1250) — matrix operand order repaired and checked by reading, not compiled, not run on hardware
//
#include <hip/hip_runtime.h>
#include <hip/hip_bf16.h>

typedef __bf16 bf16_t;
typedef __attribute__((ext_vector_type(16))) __bf16 v16bf;
typedef __attribute__((ext_vector_type(8)))  __bf16 v8bf;
typedef __attribute__((ext_vector_type(8)))  float  v8f;

#define S_LEN 2048
#define BATCH 16
#define DM    512
#define NH    12
#define TOK   (S_LEN * BATCH)
#define HDP   64   // padded head dim (heads are 42/43 wide)

// ---------------------------------------------------------------------------
// WMMA fragment helpers (wave32, v_wmma_f32_16x16x32_bf16)
// A-frag 16x32 (MxK): lane<16 holds row M=lane&15, K={kh..kh+7, 16+kh..23+kh},
// kh = (lane>=16)*8. Contiguous 8-element (16B) chunks -> b128 loads.
// ---------------------------------------------------------------------------
__device__ __forceinline__ v16bf load_a_frag(const bf16_t* A, int ld, int row0, int k0) {
  int lane = threadIdx.x & 31;
  int m    = lane & 15;
  int kh   = (lane >> 4) << 3;          // 0 or 8
  const bf16_t* p = A + (size_t)(row0 + m) * ld + k0 + kh;
  v8bf lo = *(const v8bf*)p;            // K = k0+kh   .. +7
  v8bf hi = *(const v8bf*)(p + 16);     // K = k0+kh+16.. +23
  return __builtin_shufflevector(lo, hi, 0,1,2,3,4,5,6,7,8,9,10,11,12,13,14,15);
}

// B-frag 32x16 (KxN) read from a ROW-MAJOR TRANSPOSED matrix BT[N][K]:
// lane holds column N=lane&15, K={koff..koff+7, 16+koff..23+koff} (koff=(lane>=16)*8),
// i.e. two 8-element (16B) chunks per lane, same K order as the A-frag.
__device__ __forceinline__ v16bf load_bT_frag(const bf16_t* BT, int ld, int col0, int k0) {
  int lane = threadIdx.x & 31;
  int n    = lane & 15;
  int koff = (lane >> 4) << 3;          // 0 or 8
  const bf16_t* p = BT + (size_t)(col0 + n) * ld + k0 + koff;
  v8bf lo = *(const v8bf*)p;
  v8bf hi = *(const v8bf*)(p + 16);
  return __builtin_shufflevector(lo, hi, 0,1,2,3,4,5,6,7,8,9,10,11,12,13,14,15);
}

__device__ __forceinline__ v8f wmma_bf16(v16bf a, v16bf b, v8f c) {
  return __builtin_amdgcn_wmma_f32_16x16x32_bf16(false, a, false, b, (short)0, c, false, false);
}

// Dense 32x64 wave tile, K=512: two A-frags share four staged B-frags
// (1.5 b128 loads per WMMA, distinct live ranges so loads pipeline).
__device__ __forceinline__ void gemm32x64(const bf16_t* __restrict__ A, int lda,
                                          const bf16_t* __restrict__ BT, int ldb,
                                          int row0, int col0, v8f acc[2][4]) {
#pragma unroll
  for (int k0 = 0; k0 < DM; k0 += 32) {
    v16bf a0 = load_a_frag(A, lda, row0, k0);
    v16bf a1 = load_a_frag(A, lda, row0 + 16, k0);
    v16bf b[4];
#pragma unroll
    for (int i = 0; i < 4; ++i) b[i] = load_bT_frag(BT, ldb, col0 + i * 16, k0);
#pragma unroll
    for (int i = 0; i < 4; ++i) {
      acc[0][i] = wmma_bf16(a0, b[i], acc[0][i]);
      acc[1][i] = wmma_bf16(a1, b[i], acc[1][i]);
    }
  }
}

// map a 0..511 model column to (head, offset-in-head, head_dim)
__device__ __forceinline__ void col_to_head(int col, int& h, int& off, int& hd) {
  if (col < 8 * 43) { h = col / 43; off = col - h * 43; hd = 43; }
  else { int c = col - 344; h = 8 + c / 42; off = c - (h - 8) * 42; hd = 42; }
}

// ---------------------------------------------------------------------------
// K0: convert + transpose the 5 weight matrices (512x512 fp32 -> WT bf16)
// ---------------------------------------------------------------------------
__global__ void convert_w_kernel(const float* __restrict__ Wq, const float* __restrict__ Wk,
                                 const float* __restrict__ Wv, const float* __restrict__ W1,
                                 const float* __restrict__ W2, bf16_t* __restrict__ WT) {
  int z = blockIdx.z;
  const float* W = (z == 0) ? Wq : (z == 1) ? Wk : (z == 2) ? Wv : (z == 3) ? W1 : W2;
  int idx = blockIdx.x * 256 + threadIdx.x;     // idx = k*512 + n
  int k = idx >> 9, n = idx & 511;
  WT[(size_t)z * DM * DM + (size_t)n * DM + k] = (bf16_t)W[idx];
}

// ---------------------------------------------------------------------------
// K1/K4: LayerNorm, one wave per token (512 floats -> 16 per lane)
// ---------------------------------------------------------------------------
__global__ void ln_kernel(const float* __restrict__ xin, const float* __restrict__ resid,
                          const float* __restrict__ gamma, const float* __restrict__ beta,
                          float* __restrict__ out_f32, bf16_t* __restrict__ out_bf) {
  int wave = threadIdx.x >> 5, lane = threadIdx.x & 31;
  size_t t = (size_t)blockIdx.x * 8 + wave;
  const float* row  = xin + t * DM;
  const float* rrow = resid ? resid + t * DM : nullptr;
  float vals[16];
  float s = 0.f;
#pragma unroll
  for (int j = 0; j < 4; ++j) {
    int c0 = (lane + 32 * j) * 4;
#pragma unroll
    for (int c = 0; c < 4; ++c) {
      float v = row[c0 + c];
      if (rrow) v += rrow[c0 + c];
      vals[j * 4 + c] = v;
      s += v;
    }
  }
#pragma unroll
  for (int m = 1; m < 32; m <<= 1) s += __shfl_xor(s, m, 32);
  float mu = s * (1.0f / DM);
  float vv = 0.f;
#pragma unroll
  for (int e = 0; e < 16; ++e) { float d = vals[e] - mu; vv += d * d; }
#pragma unroll
  for (int m = 1; m < 32; m <<= 1) vv += __shfl_xor(vv, m, 32);
  float rstd = rsqrtf(vv * (1.0f / DM) + 1e-5f);
#pragma unroll
  for (int j = 0; j < 4; ++j) {
    int c0 = (lane + 32 * j) * 4;
#pragma unroll
    for (int c = 0; c < 4; ++c) {
      int col = c0 + c;
      float o = (vals[j * 4 + c] - mu) * rstd * gamma[col] + beta[col];
      if (out_f32) out_f32[t * DM + col] = o;
      out_bf[t * DM + col] = (bf16_t)o;
    }
  }
}

// ---------------------------------------------------------------------------
// K2: QKV projection GEMM. wave -> 32x64 tile. Epilogue scatters into padded
// per-head layouts: Q,K : [b][h][s][64] bf16 (Q pre-scaled by 1/sqrt(hd)),
// V : [b][h][64][s] bf16 (transposed so PV B-frags are contiguous).
// ---------------------------------------------------------------------------
__global__ void qkv_gemm_kernel(const bf16_t* __restrict__ A,
                                const bf16_t* __restrict__ WqT, const bf16_t* __restrict__ WkT,
                                const bf16_t* __restrict__ WvT,
                                bf16_t* __restrict__ Qp, bf16_t* __restrict__ Kp,
                                bf16_t* __restrict__ Vt) {
  int wave = threadIdx.x >> 5, lane = threadIdx.x & 31;
  int row0 = (blockIdx.x * 4 + wave) * 32;
  int col0 = blockIdx.y * 64;
  int mode = blockIdx.z;                 // 0=Q 1=K 2=V
  const bf16_t* WT = (mode == 0) ? WqT : (mode == 1) ? WkT : WvT;

  v8f z8 = {0.f,0.f,0.f,0.f,0.f,0.f,0.f,0.f};
  v8f acc[2][4] = {{z8, z8, z8, z8}, {z8, z8, z8, z8}};
  gemm32x64(A, DM, WT, DM, row0, col0, acc);

  int rbase = (lane >> 4) << 3;
  int nn    = lane & 15;
#pragma unroll
  for (int i = 0; i < 4; ++i) {
    int col = col0 + i * 16 + nn;
    int h, off, hd;
    col_to_head(col, h, off, hd);
    float scale = (mode == 0) ? rsqrtf((float)hd) : 1.0f;
#pragma unroll
    for (int half = 0; half < 2; ++half) {
#pragma unroll
      for (int r = 0; r < 8; ++r) {
        int t  = row0 + half * 16 + rbase + r;
        int sq = t >> 4, b = t & 15;       // x is (S,B,D): t = s*16 + b
        float v = acc[half][i][r] * scale;
        if (mode == 2)
          Vt[(((size_t)b * NH + h) * HDP + off) * S_LEN + sq] = (bf16_t)v;
        else {
          bf16_t* dst = (mode == 0) ? Qp : Kp;
          dst[(((size_t)b * NH + h) * S_LEN + sq) * HDP + off] = (bf16_t)v;
        }
      }
    }
  }
}

// ---------------------------------------------------------------------------
// K3: flash attention per (batch, head). Block = 4 waves x 16 query rows.
// Streams 64 keys/iter: 8 WMMA scores (B-frags staged), online softmax via
// 16-lane shfl_xor butterflies, P staged through LDS (s_wait_dscnt) and
// re-read as A-fragments, 8 WMMA for PV (B-frags staged).
// ---------------------------------------------------------------------------
__global__ void attn_kernel(const bf16_t* __restrict__ Qp, const bf16_t* __restrict__ Kp,
                            const bf16_t* __restrict__ Vt, float* __restrict__ attn_out) {
  __shared__ __align__(16) bf16_t plds[4][16][72];   // per-wave 16x64 P tile, padded stride
  int b = blockIdx.z, h = blockIdx.y;
  int wave = threadIdx.x >> 5, lane = threadIdx.x & 31;
  int row0 = blockIdx.x * 64 + wave * 16;

  const bf16_t* Q = Qp + (((size_t)b * NH + h) * S_LEN) * HDP;
  const bf16_t* K = Kp + (((size_t)b * NH + h) * S_LEN) * HDP;
  const bf16_t* V = Vt + (((size_t)b * NH + h) * HDP) * S_LEN;

  // Q fragments live in registers for the whole key loop (K-dim = 64 padded)
  v16bf qa0 = load_a_frag(Q, HDP, row0, 0);
  v16bf qa1 = load_a_frag(Q, HDP, row0, 32);

  v8f z8 = {0.f,0.f,0.f,0.f,0.f,0.f,0.f,0.f};
  v8f o[4] = {z8, z8, z8, z8};
  float mrow[8], lrow[8];
#pragma unroll
  for (int r = 0; r < 8; ++r) { mrow[r] = -3.0e38f; lrow[r] = 0.f; }
  int rbase = (lane >> 4) << 3;

  for (int kb = 0; kb < S_LEN; kb += 64) {
    // scores S = Q * K^T  (16 queries x 64 keys); stage all 8 B-frags first
    v16bf kb0[4], kb1[4];
#pragma unroll
    for (int i = 0; i < 4; ++i) {
      kb0[i] = load_bT_frag(K, HDP, kb + i * 16, 0);
      kb1[i] = load_bT_frag(K, HDP, kb + i * 16, 32);
    }
    v8f sc[4] = {z8, z8, z8, z8};
#pragma unroll
    for (int i = 0; i < 4; ++i) {
      sc[i] = wmma_bf16(qa0, kb0[i], sc[i]);
      sc[i] = wmma_bf16(qa1, kb1[i], sc[i]);
    }
    // online softmax; acc element r belongs to query row rbase+r,
    // row spread over 16 lanes (xor 1,2,4,8 stays inside the half)
#pragma unroll
    for (int r = 0; r < 8; ++r) {
      float mx = fmaxf(fmaxf(sc[0][r], sc[1][r]), fmaxf(sc[2][r], sc[3][r]));
#pragma unroll
      for (int m = 1; m < 16; m <<= 1) mx = fmaxf(mx, __shfl_xor(mx, m, 32));
      float mnew = fmaxf(mrow[r], mx);
      float corr = __expf(mrow[r] - mnew);
      mrow[r] = mnew;
      lrow[r] *= corr;
      float psum = 0.f;
#pragma unroll
      for (int i = 0; i < 4; ++i) {
        o[i][r] *= corr;
        float p = __expf(sc[i][r] - mnew);
        psum += p;
        plds[wave][rbase + r][i * 16 + (lane & 15)] = (bf16_t)p;
      }
#pragma unroll
      for (int m = 1; m < 16; m <<= 1) psum += __shfl_xor(psum, m, 32);
      lrow[r] += psum;
    }
    asm volatile("s_wait_dscnt 0x0" ::: "memory");   // LDS P tile visible to own wave
    // O += P * V   (K-dim = 64 keys, two 32-chunks); stage V B-frags
#pragma unroll
    for (int c = 0; c < 2; ++c) {
      v16bf vb[4];
#pragma unroll
      for (int i = 0; i < 4; ++i) vb[i] = load_bT_frag(V, S_LEN, i * 16, kb + c * 32);
      v16bf pa = load_a_frag(&plds[wave][0][0], 72, 0, c * 32);
#pragma unroll
      for (int i = 0; i < 4; ++i) o[i] = wmma_bf16(pa, vb[i], o[i]);
    }
  }

  int hd     = (h < 8) ? 43 : 42;
  int hstart = (h < 8) ? 43 * h : 344 + 42 * (h - 8);
#pragma unroll
  for (int i = 0; i < 4; ++i) {
    int cl = i * 16 + (lane & 15);
    if (cl < hd) {
#pragma unroll
      for (int r = 0; r < 8; ++r) {
        int sq = row0 + rbase + r;
        size_t t = (size_t)sq * BATCH + b;
        attn_out[t * DM + hstart + cl] = o[i][r] / lrow[r];
      }
    }
  }
}

// ---------------------------------------------------------------------------
// K5: H1 = relu(attn_ln @ W1^T + b1) -> bf16
// ---------------------------------------------------------------------------
__global__ void ffn1_kernel(const bf16_t* __restrict__ A, const bf16_t* __restrict__ W1T,
                            const float* __restrict__ bias, bf16_t* __restrict__ H1) {
  int wave = threadIdx.x >> 5, lane = threadIdx.x & 31;
  int row0 = (blockIdx.x * 4 + wave) * 32;
  int col0 = blockIdx.y * 64;
  v8f z8 = {0.f,0.f,0.f,0.f,0.f,0.f,0.f,0.f};
  v8f acc[2][4] = {{z8, z8, z8, z8}, {z8, z8, z8, z8}};
  gemm32x64(A, DM, W1T, DM, row0, col0, acc);

  int rbase = (lane >> 4) << 3, nn = lane & 15;
#pragma unroll
  for (int i = 0; i < 4; ++i) {
    int col = col0 + i * 16 + nn;
    float bb = bias[col];
#pragma unroll
    for (int half = 0; half < 2; ++half) {
#pragma unroll
      for (int r = 0; r < 8; ++r) {
        size_t t = (size_t)row0 + half * 16 + rbase + r;
        float v = fmaxf(acc[half][i][r] + bb, 0.f);
        H1[t * DM + col] = (bf16_t)v;
      }
    }
  }
}

// ---------------------------------------------------------------------------
// K6: out = attn_ln + (H1 @ W2^T + b2) -> fp32 d_out
// ---------------------------------------------------------------------------
__global__ void ffn2_kernel(const bf16_t* __restrict__ A, const bf16_t* __restrict__ W2T,
                            const float* __restrict__ bias, const float* __restrict__ resid,
                            float* __restrict__ out) {
  int wave = threadIdx.x >> 5, lane = threadIdx.x & 31;
  int row0 = (blockIdx.x * 4 + wave) * 32;
  int col0 = blockIdx.y * 64;
  v8f z8 = {0.f,0.f,0.f,0.f,0.f,0.f,0.f,0.f};
  v8f acc[2][4] = {{z8, z8, z8, z8}, {z8, z8, z8, z8}};
  gemm32x64(A, DM, W2T, DM, row0, col0, acc);

  int rbase = (lane >> 4) << 3, nn = lane & 15;
#pragma unroll
  for (int i = 0; i < 4; ++i) {
    int col = col0 + i * 16 + nn;
    float bb = bias[col];
#pragma unroll
    for (int half = 0; half < 2; ++half) {
#pragma unroll
      for (int r = 0; r < 8; ++r) {
        size_t t = (size_t)row0 + half * 16 + rbase + r;
        out[t * DM + col] = acc[half][i][r] + bb + resid[t * DM + col];
      }
    }
  }
}

// ---------------------------------------------------------------------------
extern "C" void kernel_launch(void* const* d_in, const int* in_sizes, int n_in,
                              void* d_out, int out_size, void* d_ws, size_t ws_size,
                              hipStream_t stream) {
  (void)in_sizes; (void)n_in; (void)out_size; (void)ws_size;
  const float* x   = (const float*)d_in[0];
  const float* Wq  = (const float*)d_in[1];
  const float* Wk  = (const float*)d_in[2];
  const float* Wv  = (const float*)d_in[3];
  const float* g1  = (const float*)d_in[4];
  const float* b1  = (const float*)d_in[5];
  const float* g2  = (const float*)d_in[6];
  const float* b2  = (const float*)d_in[7];
  const float* W1  = (const float*)d_in[8];
  const float* bf1 = (const float*)d_in[9];
  const float* W2  = (const float*)d_in[10];
  const float* bf2 = (const float*)d_in[11];

  char* w = (char*)d_ws;
  size_t off = 0;
  bf16_t* normed  = (bf16_t*)(w + off); off += (size_t)TOK * DM * 2;
  bf16_t* WT      = (bf16_t*)(w + off); off += 5ull * DM * DM * 2;
  bf16_t* Qp      = (bf16_t*)(w + off); off += (size_t)BATCH * NH * S_LEN * HDP * 2;
  bf16_t* Kp      = (bf16_t*)(w + off); off += (size_t)BATCH * NH * S_LEN * HDP * 2;
  bf16_t* Vt      = (bf16_t*)(w + off); off += (size_t)BATCH * NH * HDP * S_LEN * 2;
  float*  attn_c  = (float*)(w + off);  off += (size_t)TOK * DM * 4;
  float*  attn_ln = (float*)(w + off);  off += (size_t)TOK * DM * 4;
  bf16_t* attn_lb = (bf16_t*)(w + off); off += (size_t)TOK * DM * 2;
  bf16_t* h1      = (bf16_t*)(w + off); off += (size_t)TOK * DM * 2;

  bf16_t* WqT = WT + 0ull * DM * DM;
  bf16_t* WkT = WT + 1ull * DM * DM;
  bf16_t* WvT = WT + 2ull * DM * DM;
  bf16_t* W1T = WT + 3ull * DM * DM;
  bf16_t* W2T = WT + 4ull * DM * DM;

  // K0: weights -> bf16 transposed
  convert_w_kernel<<<dim3(DM * DM / 256, 1, 5), 256, 0, stream>>>(Wq, Wk, Wv, W1, W2, WT);
  // K1: LayerNorm1 -> bf16
  ln_kernel<<<TOK / 8, 256, 0, stream>>>(x, nullptr, g1, b1, nullptr, normed);
  // zero the padded per-head Q/K/V region (capturable async memset)
  hipMemsetAsync(Qp, 0, 3ull * BATCH * NH * S_LEN * HDP * 2, stream);
  // K2: QKV projections into padded head layouts (32-row wave tiles)
  qkv_gemm_kernel<<<dim3(TOK / 128, DM / 64, 3), 128, 0, stream>>>(normed, WqT, WkT, WvT, Qp, Kp, Vt);
  // K3: flash attention per (b, h)
  attn_kernel<<<dim3(S_LEN / 64, NH, BATCH), 128, 0, stream>>>(Qp, Kp, Vt, attn_c);
  // K4: residual + LayerNorm2 -> fp32 + bf16
  ln_kernel<<<TOK / 8, 256, 0, stream>>>(attn_c, x, g2, b2, attn_ln, attn_lb);
  // K5/K6: FFN
  ffn1_kernel<<<dim3(TOK / 128, DM / 64), 128, 0, stream>>>(attn_lb, W1T, bf1, h1);
  ffn2_kernel<<<dim3(TOK / 128, DM / 64), 128, 0, stream>>>(h1, W2T, bf2, attn_ln, (float*)d_out);
}
